// VarianceAdaptor_82652350644367
// MI455X (gfx1250) — compile-verified, matrix operand order checked by reading, not run on hardware
//
#include <hip/hip_runtime.h>

// ---------------------------------------------------------------------------
// VarianceAdaptor on MI455X (gfx1250): bf16 WMMA conv-GEMMs + fused epilogues.
// Dims: B=32, T=1024, D=H=512, NBINS=256, MAX_MEL=4096.
// ---------------------------------------------------------------------------

typedef __attribute__((ext_vector_type(16))) __bf16 v16bf;
typedef __attribute__((ext_vector_type(8)))  float  v8f;

struct alignas(16) U4  { unsigned int x, y, z, w; };
struct alignas(8)  US4 { unsigned short a, b, c, d; };

union AF { U4 q[2]; v16bf v; };   // 32 bytes: one 16x32-bf16 WMMA operand per lane

__device__ __forceinline__ unsigned short f2b(float f) {
    unsigned u = __float_as_uint(f);
    unsigned r = u + 0x7FFFu + ((u >> 16) & 1u);   // round-to-nearest-even
    return (unsigned short)(r >> 16);
}
__device__ __forceinline__ float b2f(unsigned short b) {
    return __uint_as_float(((unsigned)b) << 16);
}

// ------------------------- fp32 -> bf16 convert ----------------------------
__global__ __launch_bounds__(256) void cvt_f32_bf16(const float* __restrict__ src,
                                                    unsigned short* __restrict__ dst,
                                                    int n4) {
    int i = blockIdx.x * 256 + threadIdx.x;
    if (i < n4) {
        float4 v = reinterpret_cast<const float4*>(src)[i];
        US4 o = { f2b(v.x), f2b(v.y), f2b(v.z), f2b(v.w) };
        reinterpret_cast<US4*>(dst)[i] = o;
    }
}

// --------------- conv weights (3,512,512)=[K][N] -> Wt[N][K] bf16 ----------
__global__ __launch_bounds__(256) void wt_convert(const float* __restrict__ w,
                                                  unsigned short* __restrict__ Wt) {
    int i = blockIdx.x * 256 + threadIdx.x;     // 0 .. 1536*512-1
    if (i < 1536 * 512) {
        int K = i >> 9, n = i & 511;
        Wt[(size_t)n * 1536 + K] = f2b(w[i]);
    }
}

// ------------------ WMMA conv-as-GEMM: C = relu(A*W + b) -------------------
// A: (B*T, 512) bf16, conv taps folded into K (K = kk*512 + d, time shift kk-1)
// Wt: (512, 1536) bf16 row-major [n][k].  C out: (B*T, 512) f32.
#define TM 128
#define TN 128
#define KB 32
#define LDA 40

__global__ __launch_bounds__(256) void gemm_conv(const unsigned short* __restrict__ A,
                                                 const unsigned short* __restrict__ Wt,
                                                 const float* __restrict__ bias,
                                                 float* __restrict__ Cout) {
    __shared__ alignas(16) unsigned short sa[TM * LDA];
    __shared__ alignas(16) unsigned short sb[TN * LDA];

    const int tid  = threadIdx.x;
    const int wid  = tid >> 5;
    const int lane = tid & 31;
    const int lm   = lane & 15;
    const int half = lane >> 4;
    const int m0 = blockIdx.x * TM;
    const int n0 = blockIdx.y * TN;
    const int wm = (wid & 3) * 32;    // 4 waves along M
    const int wn = (wid >> 2) * 64;   // 2 waves along N

    const int sr = tid >> 1;          // staging row 0..127
    const int sh = tid & 1;           // which 16-element half of the 32-wide K slab

    v8f acc[2][4];
#pragma unroll
    for (int i = 0; i < 2; ++i)
#pragma unroll
        for (int j = 0; j < 4; ++j)
            acc[i][j] = (v8f){0.f,0.f,0.f,0.f,0.f,0.f,0.f,0.f};

    const int mrow = m0 + sr;
    const int bidx = mrow >> 10;      // batch
    const int trow = mrow & 1023;     // time

    for (int k0 = 0; k0 < 1536; k0 += KB) {
        const int kk    = k0 >> 9;    // conv tap 0..2 -> time shift kk-1
        const int dbase = k0 & 511;

        { // stage A tile (conv shifted rows, zero padded at sequence edges)
            int tt = trow + kk - 1;
            U4 q0 = {0,0,0,0}, q1 = {0,0,0,0};
            if ((unsigned)tt < 1024u) {
                const U4* src = reinterpret_cast<const U4*>(
                    A + ((size_t)((bidx << 10) | tt) * 512 + dbase + sh * 16));
                q0 = src[0]; q1 = src[1];
            }
            U4* dst = reinterpret_cast<U4*>(&sa[sr * LDA + sh * 16]);
            dst[0] = q0; dst[1] = q1;
        }
        { // stage B tile (rows of transposed weights: contiguous in K)
            const U4* src = reinterpret_cast<const U4*>(
                Wt + (size_t)(n0 + sr) * 1536 + k0 + sh * 16);
            U4* dst = reinterpret_cast<U4*>(&sb[sr * LDA + sh * 16]);
            dst[0] = src[0]; dst[1] = src[1];
        }
        if (k0 + KB < 1536) { // prefetch next K slab -> global_prefetch_b8
            int nkk = (k0 + KB) >> 9, ndb = (k0 + KB) & 511;
            int ntt = trow + nkk - 1;
            if ((unsigned)ntt < 1024u)
                __builtin_prefetch(A + ((size_t)((bidx << 10) | ntt) * 512 + ndb + sh * 16), 0, 1);
            __builtin_prefetch(Wt + (size_t)(n0 + sr) * 1536 + (k0 + KB) + sh * 16, 0, 1);
        }
        __syncthreads();

        // Per-lane fragment loads (ISA 16-bit 16x32 layout:
        // lanes 0-15: K {0..7,16..23}; lanes 16-31: K {8..15,24..31})
        AF af[2], bf[4];
#pragma unroll
        for (int i = 0; i < 2; ++i) {
            const int r = wm + i * 16 + lm;
            af[i].q[0] = *reinterpret_cast<const U4*>(&sa[r * LDA + half * 8]);
            af[i].q[1] = *reinterpret_cast<const U4*>(&sa[r * LDA + 16 + half * 8]);
        }
#pragma unroll
        for (int j = 0; j < 4; ++j) {
            const int r = wn + j * 16 + lm;
            bf[j].q[0] = *reinterpret_cast<const U4*>(&sb[r * LDA + half * 8]);
            bf[j].q[1] = *reinterpret_cast<const U4*>(&sb[r * LDA + 16 + half * 8]);
        }
#pragma unroll
        for (int i = 0; i < 2; ++i)
#pragma unroll
            for (int j = 0; j < 4; ++j)
                acc[i][j] = __builtin_amdgcn_wmma_f32_16x16x32_bf16(
                    false, af[i].v, false, bf[j].v, (short)0, acc[i][j], false, false);
        __syncthreads();
    }

    // Epilogue: bias + ReLU, f32 store. C/D layout: lane<16 -> M=v, lane>=16 -> M=v+8.
#pragma unroll
    for (int i = 0; i < 2; ++i) {
        const int row_base = m0 + wm + i * 16 + half * 8;
#pragma unroll
        for (int j = 0; j < 4; ++j) {
            const int col = n0 + wn + j * 16 + lm;
            const float bv = bias[col];
#pragma unroll
            for (int v = 0; v < 8; ++v) {
                float val = acc[i][j][v] + bv;
                val = val > 0.f ? val : 0.f;
                Cout[(size_t)(row_base + v) * 512 + col] = val;
            }
        }
    }
}

// ------------------------------ LayerNorm ----------------------------------
__global__ __launch_bounds__(128) void ln_kernel(const float* __restrict__ C,
                                                 const float* __restrict__ g,
                                                 const float* __restrict__ be,
                                                 unsigned short* __restrict__ out) {
    __shared__ float red[128];
    const int row = blockIdx.x, tid = threadIdx.x;
    const float* x = C + (size_t)row * 512;
    float v[4], s = 0.f;
#pragma unroll
    for (int i = 0; i < 4; ++i) { v[i] = x[tid + 128 * i]; s += v[i]; }
    red[tid] = s; __syncthreads();
    for (int off = 64; off > 0; off >>= 1) { if (tid < off) red[tid] += red[tid + off]; __syncthreads(); }
    const float mu = red[0] * (1.f / 512.f);
    __syncthreads();
    float q = 0.f;
#pragma unroll
    for (int i = 0; i < 4; ++i) { float d = v[i] - mu; q += d * d; }
    red[tid] = q; __syncthreads();
    for (int off = 64; off > 0; off >>= 1) { if (tid < off) red[tid] += red[tid + off]; __syncthreads(); }
    const float rs = rsqrtf(red[0] * (1.f / 512.f) + 1e-5f);
#pragma unroll
    for (int i = 0; i < 4; ++i) {
        int d = tid + 128 * i;
        out[(size_t)row * 512 + d] = f2b((v[i] - mu) * rs * g[d] + be[d]);
    }
}

// ------------------------- final projection H->1 ---------------------------
__global__ __launch_bounds__(256) void final_proj(const unsigned short* __restrict__ h,
                                                  const float* __restrict__ wl,
                                                  const float* __restrict__ bl,
                                                  const unsigned char* __restrict__ msk,
                                                  float* __restrict__ pred) {
    const int wid = threadIdx.x >> 5, lane = threadIdx.x & 31;
    const int row = blockIdx.x * 8 + wid;   // grid=4096 -> rows 0..32767
    const unsigned short* x = h + (size_t)row * 512;
    float s = 0.f;
#pragma unroll
    for (int e = 0; e < 16; ++e) { int d = e * 32 + lane; s += b2f(x[d]) * wl[d]; }
    for (int off = 16; off > 0; off >>= 1) s += __shfl_xor(s, off, 32);
    if (lane == 0) pred[row] = msk[row] ? 0.f : (s + bl[0]);
}

// -------- bucketize (closed-form searchsorted on linspace) + embed add -----
__device__ __forceinline__ int bucketize(float v, float mn, float mx) {
    float step = (mx - mn) * (1.f / 254.f);   // NBINS-1 = 255 bins edges, 254 gaps
    int i = (int)ceilf((v - mn) / step);
    return i < 0 ? 0 : (i > 255 ? 255 : i);
}

__global__ __launch_bounds__(128) void combine_kernel(const float* __restrict__ batch,
                                                      const float* __restrict__ pp,
                                                      const float* __restrict__ ep,
                                                      const float* __restrict__ pe_emb,
                                                      const float* __restrict__ ee_emb,
                                                      float* __restrict__ X,
                                                      unsigned short* __restrict__ Xb) {
    const int row = blockIdx.x, tid = threadIdx.x;
    const int ip = bucketize(pp[row], -3.f, 12.f);
    const int ie = bucketize(ep[row], -2.f, 9.f);
#pragma unroll
    for (int i = 0; i < 4; ++i) {
        int d = tid + 128 * i;
        float v = batch[(size_t)row * 512 + d] + pe_emb[(size_t)ip * 512 + d]
                                               + ee_emb[(size_t)ie * 512 + d];
        X[(size_t)row * 512 + d]  = v;
        Xb[(size_t)row * 512 + d] = f2b(v);
    }
}

// --------------------- duration -> inclusive scan per batch ----------------
__global__ __launch_bounds__(1024) void scan_kernel(const float* __restrict__ raw,
                                                    const unsigned char* __restrict__ msk,
                                                    int* __restrict__ cum) {
    __shared__ int s[1024];
    const int b = blockIdx.x, t = threadIdx.x;
    int d = 0;
    if (!msk[b * 1024 + t]) {
        float e = expf(raw[b * 1024 + t]) - 1.f;
        int r = (int)rintf(e);
        d = r > 0 ? r : 0;
    }
    s[t] = d; __syncthreads();
    for (int off = 1; off < 1024; off <<= 1) {
        int v = (t >= off) ? s[t - off] : 0;
        __syncthreads();
        s[t] += v;
        __syncthreads();
    }
    cum[b * 1024 + t] = s[t];
}

// --------------- length regulator: searchsorted(right) + gather ------------
__global__ __launch_bounds__(128) void gather_kernel(const float* __restrict__ X,
                                                     const int* __restrict__ cum,
                                                     float* __restrict__ out,
                                                     float* __restrict__ om) {
    const int pos = blockIdx.x, b = blockIdx.y, tid = threadIdx.x;
    const int* c = cum + b * 1024;
    const int total = c[1023];
    const size_t obase = ((size_t)b * 4096 + pos) * 512;
    if (pos >= total) {
        float4 z = {0.f, 0.f, 0.f, 0.f};
        reinterpret_cast<float4*>(out + obase)[tid] = z;
        if (tid == 0) om[b * 4096 + pos] = 1.f;
        return;
    }
    int lo = 0, hi = 1024;                 // first i with c[i] > pos (side='right')
    while (lo < hi) { int mid = (lo + hi) >> 1; if (c[mid] > pos) hi = mid; else lo = mid + 1; }
    int src = lo > 1023 ? 1023 : lo;
    reinterpret_cast<float4*>(out + obase)[tid] =
        reinterpret_cast<const float4*>(X + ((size_t)b * 1024 + src) * 512)[tid];
    if (tid == 0) om[b * 4096 + pos] = 0.f;
}

// ---------------------------------------------------------------------------
static void run_predictor(const unsigned short* inA, const float* const* P,
                          unsigned short* A1, float* C, unsigned short* Wt,
                          const unsigned char* msk, float* predOut, hipStream_t s) {
    const float *w1 = P[0], *b1 = P[1], *g1 = P[2], *be1 = P[3],
                *w2 = P[4], *b2 = P[5], *g2 = P[6], *be2 = P[7],
                *wl = P[8], *bl = P[9];
    dim3 gg(256, 4);
    wt_convert<<<3072, 256, 0, s>>>(w1, Wt);
    gemm_conv<<<gg, 256, 0, s>>>(inA, Wt, b1, C);
    ln_kernel<<<32768, 128, 0, s>>>(C, g1, be1, A1);
    wt_convert<<<3072, 256, 0, s>>>(w2, Wt);
    gemm_conv<<<gg, 256, 0, s>>>(A1, Wt, b2, C);
    ln_kernel<<<32768, 128, 0, s>>>(C, g2, be2, A1);
    final_proj<<<4096, 256, 0, s>>>(A1, wl, bl, msk, predOut);
}

extern "C" void kernel_launch(void* const* d_in, const int* in_sizes, int n_in,
                              void* d_out, int out_size, void* d_ws, size_t ws_size,
                              hipStream_t stream) {
    (void)in_sizes; (void)n_in; (void)out_size; (void)ws_size;

    const float*         batch = (const float*)d_in[0];
    const unsigned char* msk   = (const unsigned char*)d_in[1];
    const float* pitchP[10], *energyP[10], *durP[10];
    for (int i = 0; i < 10; ++i) {
        pitchP[i]  = (const float*)d_in[2 + i];
        energyP[i] = (const float*)d_in[12 + i];
        durP[i]    = (const float*)d_in[22 + i];
    }
    const float* pe_emb = (const float*)d_in[32];
    const float* ee_emb = (const float*)d_in[33];

    // d_out layout (floats): out | pitch_pred | energy_pred | raw_dur | out_mask
    float* out = (float*)d_out;
    const size_t OUT_E = 32ull * 4096 * 512;        // 67,108,864
    float* pp = out + OUT_E;
    float* ep = pp + 32768;
    float* rd = ep + 32768;
    float* om = rd + 32768;

    // workspace layout
    char* ws = (char*)d_ws;
    unsigned short* A0 = (unsigned short*)(ws);                 // 33,554,432 B
    unsigned short* A1 = (unsigned short*)(ws + 33554432);      // 33,554,432 B
    float*          C  = (float*)        (ws + 67108864);       // 67,108,864 B
    unsigned short* Wt = (unsigned short*)(ws + 134217728);     //  1,572,864 B
    float*          X  = (float*)        (ws + 135790592);      // 67,108,864 B
    int*          cum  = (int*)          (ws + 202899456);      //    131,072 B

    // 1) batch fp32 -> bf16
    cvt_f32_bf16<<<16384, 256, 0, stream>>>(batch, A0, 32 * 1024 * 512 / 4);

    // 2) pitch + energy predictors (WMMA conv-GEMMs)
    run_predictor(A0, pitchP,  A1, C, Wt, msk, pp, stream);
    run_predictor(A0, energyP, A1, C, Wt, msk, ep, stream);

    // 3) bucketize + embedding add -> x (f32 + bf16)
    combine_kernel<<<32768, 128, 0, stream>>>(batch, pp, ep, pe_emb, ee_emb, X, A0);

    // 4) duration predictor on x
    run_predictor(A0, durP, A1, C, Wt, msk, rd, stream);

    // 5) durations -> inclusive cumsum
    scan_kernel<<<32, 1024, 0, stream>>>(rd, msk, cum);

    // 6) length regulation: searchsorted + row gather + mask
    dim3 gG(4096, 32);
    gather_kernel<<<gG, 128, 0, stream>>>(X, cum, out, om);
}
